// Attention_790273983058
// MI455X (gfx1250) — compile-verified, matrix operand order checked by reading, not run on hardware
//
#include <hip/hip_runtime.h>

// ---------------------------------------------------------------------------
// Cross-attention transformer block for MI455X (gfx1250, wave32, WMMA).
//   out0 = x3 + FFN(x3),  out1 = yn      (both fp32)
// All GEMMs run on v_wmma_f32_16x16x32_f16 (f16 A/B, f32 accum).
// Two-pass softmax: the 67MB score tensor fits in the 192MB L2.
// GEMM is NT-only (B stored [N][K] row-major): both A and B tiles move
// global->LDS with GLOBAL_LOAD_ASYNC_TO_LDS_B128 (CDNA5 async DMA, ASYNCcnt)
// when available -- no staging registers, no scratch spills. Fallback is
// register staging with relaxed VGPR cap.
// 128x128x32 workgroup tile, 8 wave32s each owning 32x64 (2x4 WMMA tiles),
// double-buffered LDS -> 1 barrier per K-step.
// ---------------------------------------------------------------------------

typedef __attribute__((ext_vector_type(16))) _Float16 v16h;
typedef __attribute__((ext_vector_type(8)))  float    v8f;

#define BATCH 4
#define SQ_   2048
#define SK_   2048
#define E_    1024
#define L_    4096

#if __has_builtin(__builtin_amdgcn_global_load_async_to_lds_b128)
#define HAVE_ASYNC_LDS 1
typedef int v4i_ __attribute__((vector_size(16)));
typedef __attribute__((address_space(1))) v4i_ gv4i;   // global int4
typedef __attribute__((address_space(3))) v4i_ lv4i;   // LDS int4

__device__ __forceinline__ void
async_copy_b128(const _Float16* g, _Float16* l)
{
    __builtin_amdgcn_global_load_async_to_lds_b128(
        (gv4i*)g, (lv4i*)l, 0, 0);
}
__device__ __forceinline__ void async_wait0()
{
#if __has_builtin(__builtin_amdgcn_s_wait_asynccnt)
    __builtin_amdgcn_s_wait_asynccnt(0);
#else
    asm volatile("s_wait_asynccnt 0" ::: "memory");
#endif
}
#else
#define HAVE_ASYNC_LDS 0
#endif

// ------------------------------- LayerNorm ---------------------------------
__global__ void __launch_bounds__(256)
ln_kernel(const float* __restrict__ in, const float* __restrict__ w,
          const float* __restrict__ b, float* __restrict__ outF,
          _Float16* __restrict__ outH)
{
    const int row = blockIdx.x;
    const int t   = threadIdx.x;
    const float* p = in + (size_t)row * E_;
    float x0 = p[t], x1 = p[t + 256], x2 = p[t + 512], x3 = p[t + 768];

    __shared__ float red[256];
    red[t] = x0 + x1 + x2 + x3;
    __syncthreads();
    for (int off = 128; off; off >>= 1) {
        if (t < off) red[t] += red[t + off];
        __syncthreads();
    }
    const float mu = red[0] * (1.0f / 1024.0f);
    __syncthreads();

    const float d0 = x0 - mu, d1 = x1 - mu, d2 = x2 - mu, d3 = x3 - mu;
    red[t] = d0 * d0 + d1 * d1 + d2 * d2 + d3 * d3;
    __syncthreads();
    for (int off = 128; off; off >>= 1) {
        if (t < off) red[t] += red[t + off];
        __syncthreads();
    }
    const float rs = rsqrtf(red[0] * (1.0f / 1024.0f) + 1e-5f);

    const float o0 = d0 * rs * w[t]       + b[t];
    const float o1 = d1 * rs * w[t + 256] + b[t + 256];
    const float o2 = d2 * rs * w[t + 512] + b[t + 512];
    const float o3 = d3 * rs * w[t + 768] + b[t + 768];

    const size_t base = (size_t)row * E_;
    outF[base + t]       = o0;
    outF[base + t + 256] = o1;
    outF[base + t + 512] = o2;
    outF[base + t + 768] = o3;
    outH[base + t]       = (_Float16)o0;
    outH[base + t + 256] = (_Float16)o1;
    outH[base + t + 512] = (_Float16)o2;
    outH[base + t + 768] = (_Float16)o3;
}

// ------------------------------- Softmax -----------------------------------
__global__ void __launch_bounds__(256)
softmax_kernel(const float* __restrict__ S, _Float16* __restrict__ P)
{
    const int row = blockIdx.x;
    const int t   = threadIdx.x;
    const float* s = S + (size_t)row * SK_;

    float v[8];
    float m = -3.4e38f;
#pragma unroll
    for (int i = 0; i < 8; i++) { v[i] = s[t + 256 * i]; m = fmaxf(m, v[i]); }

    __shared__ float red[256];
    red[t] = m;
    __syncthreads();
    for (int off = 128; off; off >>= 1) {
        if (t < off) red[t] = fmaxf(red[t], red[t + off]);
        __syncthreads();
    }
    m = red[0];
    __syncthreads();

    float sum = 0.0f;
#pragma unroll
    for (int i = 0; i < 8; i++) { v[i] = __expf(v[i] - m); sum += v[i]; }
    red[t] = sum;
    __syncthreads();
    for (int off = 128; off; off >>= 1) {
        if (t < off) red[t] += red[t + off];
        __syncthreads();
    }
    const float inv = 1.0f / red[0];

    _Float16* o = P + (size_t)row * SK_;
#pragma unroll
    for (int i = 0; i < 8; i++) o[t + 256 * i] = (_Float16)(v[i] * inv);
}

// -------------------- Tiled transpose (+convert to f16) --------------------
// in: [R][C] (TIN), out: [C][R] (f16). Batched via blockIdx.z (stride R*C).
template <typename TIN>
__global__ void __launch_bounds__(256)
transpose_cvt_kernel(const TIN* __restrict__ in, _Float16* __restrict__ out,
                     int R, int C)
{
    __shared__ _Float16 tile[64][72];   // pad to dodge bank conflicts
    const int rBase = blockIdx.y * 64;
    const int cBase = blockIdx.x * 64;
    const size_t z = blockIdx.z;
    in  += z * (size_t)R * C;
    out += z * (size_t)R * C;
    const int t = threadIdx.x;
#pragma unroll
    for (int i = 0; i < 16; i++) {
        const int idx = t + 256 * i;
        const int r = idx >> 6, c = idx & 63;
        tile[r][c] = (_Float16)in[(size_t)(rBase + r) * C + cBase + c];
    }
    __syncthreads();
#pragma unroll
    for (int i = 0; i < 16; i++) {
        const int idx  = t + 256 * i;
        const int oc   = idx & 63;     // fast dim of out row
        const int orow = idx >> 6;
        out[(size_t)(cBase + orow) * R + rBase + oc] = tile[oc][orow];
    }
}

// ---------------------- WMMA fragment load from LDS ------------------------
// 16-bit A/B fragment layout (CDNA5 ISA 7.12.2):
//   lanes 0-15 : row = lane,    K in {0..7} and {16..23}
//   lanes 16-31: row = lane-16, K in {8..15} and {24..31}
__device__ __forceinline__ v16h
load_frag(const _Float16* base, int rowBase, int pitch, int lane)
{
    const int r  = rowBase + (lane & 15);
    const int ko = (lane >> 4) << 3;
    const _Float16* p = base + r * pitch + ko;
    union { v16h v; uint4 u[2]; } f;
    f.u[0] = *(const uint4*)(p);
    f.u[1] = *(const uint4*)(p + 16);
    return f.v;
}

// ------------------------------ WMMA GEMM (NT) ------------------------------
// C[M,N] = epilogue(scale * A @ B^T), A: [M,K] row-major, B: [N,K] row-major.
enum GemmFlags {
    F_BIAS = 1,    // add bias[col]
    F_MASK = 2,    // mask!=0 -> -1e30 (applied after scale)
    F_RELU = 4,    // relu
    F_RES  = 8,    // add fp32 residual[row,col]
    F_OUTF = 16,   // write fp32 output
    F_OUTH = 32    // write f16 output
};

template <int FLAGS>
__global__ void __launch_bounds__(256)
__attribute__((amdgpu_waves_per_eu(1)))
gemm_kernel(const _Float16* __restrict__ A, const _Float16* __restrict__ Bm,
            const float* __restrict__ bias, const float* __restrict__ residual,
            const int* __restrict__ mask, float* __restrict__ outF,
            _Float16* __restrict__ outH, int M, int N, int K, float scale)
{
    constexpr bool HAS_BIAS = (FLAGS & F_BIAS) != 0;
    constexpr bool HAS_MASK = (FLAGS & F_MASK) != 0;
    constexpr bool HAS_RELU = (FLAGS & F_RELU) != 0;
    constexpr bool HAS_RES  = (FLAGS & F_RES)  != 0;
    constexpr bool OUT_F    = (FLAGS & F_OUTF) != 0;
    constexpr bool OUT_H    = (FLAGS & F_OUTH) != 0;

    constexpr int BM = 128, BN = 128, BK = 32, PITCH = 40;
    __shared__ _Float16 As[2][BM][PITCH];   // [m][k]
    __shared__ _Float16 Bs[2][BN][PITCH];   // [n][k]

    const int tid  = threadIdx.x;
    const int lane = tid & 31;
    const int wave = tid >> 5;
    const int wm   = wave & 3;    // 4 waves along M (32 rows each)
    const int wn   = wave >> 2;   // 2 waves along N (64 cols each)
    const int mBase = blockIdx.y * BM;
    const int nBase = blockIdx.x * BN;

    const size_t z = blockIdx.z;
    A  += z * (size_t)M * K;
    Bm += z * (size_t)K * N;
    const size_t cOff = z * (size_t)M * N;

    // staging indices: identical pattern for A and B (both [rows][K])
    const int sr = tid >> 2;         // 0..63 (rows sr and sr+64)
    const int sc = (tid & 3) << 3;   // 0,8,16,24

#if HAVE_ASYNC_LDS
    // ---- stage tile 0 via async DMA (ASYNCcnt) ----
    {
        const _Float16* ga = A  + (size_t)(mBase + sr) * K + sc;
        const _Float16* gb = Bm + (size_t)(nBase + sr) * K + sc;
        async_copy_b128(ga,                  &As[0][sr][sc]);
        async_copy_b128(ga + (size_t)64 * K, &As[0][sr + 64][sc]);
        async_copy_b128(gb,                  &Bs[0][sr][sc]);
        async_copy_b128(gb + (size_t)64 * K, &Bs[0][sr + 64][sc]);
    }
    async_wait0();
    __syncthreads();
#else
    uint4 areg[2], breg[2];
    // ---- stage tile 0 ----
    {
        const _Float16* ga = A  + (size_t)(mBase + sr) * K + sc;
        const _Float16* gb = Bm + (size_t)(nBase + sr) * K + sc;
        areg[0] = *(const uint4*)ga;
        areg[1] = *(const uint4*)(ga + (size_t)64 * K);
        breg[0] = *(const uint4*)gb;
        breg[1] = *(const uint4*)(gb + (size_t)64 * K);
        *(uint4*)&As[0][sr][sc]      = areg[0];
        *(uint4*)&As[0][sr + 64][sc] = areg[1];
        *(uint4*)&Bs[0][sr][sc]      = breg[0];
        *(uint4*)&Bs[0][sr + 64][sc] = breg[1];
    }
    __syncthreads();
#endif

    const v8f vzero = {0.f, 0.f, 0.f, 0.f, 0.f, 0.f, 0.f, 0.f};
    v8f acc[2][4];
#pragma unroll
    for (int i = 0; i < 2; i++)
#pragma unroll
        for (int j = 0; j < 4; j++) acc[i][j] = vzero;

    int cur = 0;
    for (int kt = 0; kt < K; kt += BK) {
        const bool hasNext = (kt + BK) < K;
        const int  nxt     = cur ^ 1;

#if HAVE_ASYNC_LDS
        // ---- async DMA of next tile directly into the other LDS buffer ----
        if (hasNext) {
            const int kn = kt + BK;
            const _Float16* ga = A  + (size_t)(mBase + sr) * K + kn + sc;
            const _Float16* gb = Bm + (size_t)(nBase + sr) * K + kn + sc;
            async_copy_b128(ga,                  &As[nxt][sr][sc]);
            async_copy_b128(ga + (size_t)64 * K, &As[nxt][sr + 64][sc]);
            async_copy_b128(gb,                  &Bs[nxt][sr][sc]);
            async_copy_b128(gb + (size_t)64 * K, &Bs[nxt][sr + 64][sc]);
        }
#else
        // ---- issue global loads for next tile (latency hidden by WMMA) ----
        if (hasNext) {
            const int kn = kt + BK;
            const _Float16* ga = A  + (size_t)(mBase + sr) * K + kn + sc;
            const _Float16* gb = Bm + (size_t)(nBase + sr) * K + kn + sc;
            areg[0] = *(const uint4*)ga;
            areg[1] = *(const uint4*)(ga + (size_t)64 * K);
            breg[0] = *(const uint4*)gb;
            breg[1] = *(const uint4*)(gb + (size_t)64 * K);
            if (kt + 2 * BK < K) {   // stream prefetch, 2 tiles ahead
                __builtin_prefetch(ga + BK, 0, 3);
                __builtin_prefetch(gb + BK, 0, 3);
            }
        }
#endif

        // ---- compute from current LDS buffer ----
        // B fragments consumed one at a time to cap VGPR pressure.
        {
            const v16h af0 = load_frag(&As[cur][0][0], wm * 32,      PITCH, lane);
            const v16h af1 = load_frag(&As[cur][0][0], wm * 32 + 16, PITCH, lane);
#pragma unroll
            for (int j = 0; j < 4; j++) {
                const v16h bf = load_frag(&Bs[cur][0][0], wn * 64 + j * 16,
                                          PITCH, lane);
                acc[0][j] = __builtin_amdgcn_wmma_f32_16x16x32_f16(
                    false, af0, false, bf, (short)0, acc[0][j], false, false);
                acc[1][j] = __builtin_amdgcn_wmma_f32_16x16x32_f16(
                    false, af1, false, bf, (short)0, acc[1][j], false, false);
            }
        }

#if HAVE_ASYNC_LDS
        if (hasNext) async_wait0();   // own async writes to nxt complete
#else
        // ---- commit staged tile into the other buffer ----
        if (hasNext) {
            *(uint4*)&As[nxt][sr][sc]      = areg[0];
            *(uint4*)&As[nxt][sr + 64][sc] = areg[1];
            *(uint4*)&Bs[nxt][sr][sc]      = breg[0];
            *(uint4*)&Bs[nxt][sr + 64][sc] = breg[1];
        }
#endif
        __syncthreads();   // next buffer visible; also fences reads of cur
        cur = nxt;
    }

    // ---- fused epilogue (straight-line; flags are compile-time) ----
    const int r8 = (lane >> 4) << 3;
    const int cl = lane & 15;
#pragma unroll
    for (int i = 0; i < 2; i++) {
#pragma unroll
        for (int j = 0; j < 4; j++) {
            const int tm  = mBase + wm * 32 + i * 16;
            const int tn  = nBase + wn * 64 + j * 16;
            const int col = tn + cl;
            float bb = 0.0f;
            if constexpr (HAS_BIAS) bb = bias[col];
#pragma unroll
            for (int r = 0; r < 8; r++) {
                const int row = tm + r8 + r;
                const size_t o = cOff + (size_t)row * N + col;
                float v = acc[i][j][r] * scale + bb;
                if constexpr (HAS_MASK) { if (mask[o] != 0) v = -1e30f; }
                if constexpr (HAS_RELU) v = fmaxf(v, 0.0f);
                if constexpr (HAS_RES)  v += residual[o];
                if constexpr (OUT_F)    outF[o] = v;
                if constexpr (OUT_H)    outH[o] = (_Float16)v;
            }
        }
    }
}

// ------------------------------- Launcher ----------------------------------
extern "C" void kernel_launch(void* const* d_in, const int* in_sizes, int n_in,
                              void* d_out, int out_size, void* d_ws,
                              size_t ws_size, hipStream_t stream)
{
    (void)in_sizes; (void)n_in; (void)out_size; (void)ws_size;

    const float* x     = (const float*)d_in[0];
    const float* y     = (const float*)d_in[1];
    const int*   mask  = (const int*)d_in[2];
    const float* ln1_w = (const float*)d_in[3];
    const float* ln1_b = (const float*)d_in[4];
    const float* ln2_w = (const float*)d_in[5];
    const float* ln2_b = (const float*)d_in[6];
    const float* ln3_w = (const float*)d_in[7];
    const float* ln3_b = (const float*)d_in[8];
    const float* Wq    = (const float*)d_in[9];
    const float* bq    = (const float*)d_in[10];
    const float* Wk    = (const float*)d_in[11];
    const float* bk    = (const float*)d_in[12];
    const float* Wv    = (const float*)d_in[13];
    const float* bv    = (const float*)d_in[14];
    const float* Win   = (const float*)d_in[15];
    const float* b_in  = (const float*)d_in[16];
    const float* Wout  = (const float*)d_in[17];
    const float* bout  = (const float*)d_in[18];

    const size_t MQ = (size_t)BATCH * SQ_;   // 8192
    const size_t MK = (size_t)BATCH * SK_;   // 8192

    float* out_main = (float*)d_out;                 // [B,SQ,E]
    float* out_yn   = (float*)d_out + MQ * E_;       // [B,SK,E]

    // -------- workspace carve (deterministic, no allocation calls) --------
    size_t cur = 0;
    char* wsb = (char*)d_ws;
    auto walloc = [&](size_t bytes) -> char* {
        char* p = wsb + cur;
        cur += (bytes + 255) & ~(size_t)255;
        return p;
    };
    float*    xnF   = (float*)   walloc(MQ * E_ * 4);
    _Float16* xnH   = (_Float16*)walloc(MQ * E_ * 2);
    _Float16* ynH   = (_Float16*)walloc(MK * E_ * 2);
    _Float16* WqT   = (_Float16*)walloc((size_t)E_ * L_ * 2);   // [L][E]
    _Float16* WkT   = (_Float16*)walloc((size_t)E_ * L_ * 2);   // [L][E]
    _Float16* WvT   = (_Float16*)walloc((size_t)E_ * E_ * 2);   // [E][E]
    _Float16* WinT  = (_Float16*)walloc((size_t)E_ * L_ * 2);   // [L][E]
    _Float16* WoutT = (_Float16*)walloc((size_t)L_ * E_ * 2);   // [E][L]
    _Float16* qH    = (_Float16*)walloc(MQ * L_ * 2);                   // 67MB
    _Float16* kH    = (_Float16*)walloc(MK * L_ * 2);                   // 67MB
    _Float16* vH    = (_Float16*)walloc(MK * E_ * 2);
    _Float16* vT    = (_Float16*)walloc(MK * E_ * 2);    // [B][E][SK]
    float*    S     = (float*)   walloc((size_t)BATCH * SQ_ * SK_ * 4); // 67MB
    float*    x3F   = (float*)   walloc(MQ * E_ * 4);
    _Float16* x3H   = (_Float16*)walloc(MQ * E_ * 2);
    // Aliases (producer runs strictly after last consumer of the old data):
    _Float16* hH  = qH;   // FFN hidden reuses q
    _Float16* PH  = kH;   // softmax probs reuse k
    float*    x2F = S;    // attention output reuses score buffer

    // -------- weight transpose+convert: W[K][N] -> f16 W^T[N][K] --------
    transpose_cvt_kernel<float><<<dim3(L_ / 64, E_ / 64, 1), 256, 0, stream>>>(Wq,   WqT,   E_, L_);
    transpose_cvt_kernel<float><<<dim3(L_ / 64, E_ / 64, 1), 256, 0, stream>>>(Wk,   WkT,   E_, L_);
    transpose_cvt_kernel<float><<<dim3(E_ / 64, E_ / 64, 1), 256, 0, stream>>>(Wv,   WvT,   E_, E_);
    transpose_cvt_kernel<float><<<dim3(L_ / 64, E_ / 64, 1), 256, 0, stream>>>(Win,  WinT,  E_, L_);
    transpose_cvt_kernel<float><<<dim3(E_ / 64, L_ / 64, 1), 256, 0, stream>>>(Wout, WoutT, L_, E_);

    // -------- layernorms --------
    ln_kernel<<<(int)MQ, 256, 0, stream>>>(x, ln1_w, ln1_b, xnF, xnH);
    ln_kernel<<<(int)MK, 256, 0, stream>>>(y, ln2_w, ln2_b, out_yn, ynH);

    constexpr int PROJ  = F_BIAS | F_OUTH;
    constexpr int SCORE = F_MASK | F_OUTF;
    constexpr int PV    = F_RES | F_OUTF;
    constexpr int FFN1  = F_BIAS | F_RELU | F_OUTH;
    constexpr int FFN2  = F_BIAS | F_RES | F_OUTF;

    // -------- projections: q, k, v --------
    gemm_kernel<PROJ><<<dim3(L_ / 128, MQ / 128, 1), 256, 0, stream>>>(
        xnH, WqT, bq, nullptr, nullptr, nullptr, qH, (int)MQ, L_, E_, 1.0f);
    gemm_kernel<PROJ><<<dim3(L_ / 128, MK / 128, 1), 256, 0, stream>>>(
        ynH, WkT, bk, nullptr, nullptr, nullptr, kH, (int)MK, L_, E_, 1.0f);
    gemm_kernel<PROJ><<<dim3(E_ / 128, MK / 128, 1), 256, 0, stream>>>(
        ynH, WvT, bv, nullptr, nullptr, nullptr, vH, (int)MK, E_, E_, 1.0f);

    // -------- transpose v per batch: [SK][E] -> [E][SK] --------
    transpose_cvt_kernel<_Float16><<<dim3(E_ / 64, SK_ / 64, BATCH), 256, 0, stream>>>(
        vH, vT, SK_, E_);

    // -------- masked scaled scores: S = q @ k^T / sqrt(L), mask -> -1e30 ---
    gemm_kernel<SCORE><<<dim3(SK_ / 128, SQ_ / 128, BATCH), 256, 0, stream>>>(
        qH, kH, nullptr, nullptr, mask, S, nullptr,
        SQ_, SK_, L_, 0.015625f /* 1/sqrt(4096) */);

    // -------- softmax rows --------
    softmax_kernel<<<BATCH * SQ_, 256, 0, stream>>>(S, PH);

    // -------- x2 = xn + P @ v   (B operand = v^T, NT layout) --------
    gemm_kernel<PV><<<dim3(E_ / 128, SQ_ / 128, BATCH), 256, 0, stream>>>(
        PH, vT, nullptr, xnF, nullptr, x2F, nullptr, SQ_, E_, SK_, 1.0f);

    // -------- ln3 --------
    ln_kernel<<<(int)MQ, 256, 0, stream>>>(x2F, ln3_w, ln3_b, x3F, x3H);

    // -------- FFN: h = relu(x3 @ Win + bin); out = x3 + h @ Wout + bout ----
    gemm_kernel<FFN1><<<dim3(L_ / 128, MQ / 128, 1), 256, 0, stream>>>(
        x3H, WinT, b_in, nullptr, nullptr, nullptr, hH, (int)MQ, L_, E_, 1.0f);
    gemm_kernel<FFN2><<<dim3(E_ / 128, MQ / 128, 1), 256, 0, stream>>>(
        hH, WoutT, bout, x3F, nullptr, out_main, nullptr, (int)MQ, E_, L_, 1.0f);
}
